// OffConv2d_89464168776167
// MI455X (gfx1250) — matrix operand order repaired and checked by reading, not compiled, not run on hardware
//
#include <hip/hip_runtime.h>
#include <hip/hip_bf16.h>

typedef __attribute__((ext_vector_type(16))) __bf16 v16bf;
typedef __attribute__((ext_vector_type(8)))  float  v8f;

constexpr int Bn = 16, Cc = 64, Hh = 128, Ww = 128;
constexpr int PLANE = Hh * Ww;                           // 16384
constexpr long long SZ = (long long)Bn * Cc * Hh * Ww;   // 16,777,216 elems
constexpr int Ksum = Cc * 9;                             // 576
constexpr int KC   = 32;                                 // WMMA K per step
constexpr int NCH  = Ksum / KC;                          // 18 chunks
constexpr int NT   = 192;                                // 3 convs x 64 ch
constexpr int BPAD = 40;                                 // B LDS row pad (bf16)
constexpr int XS2R = 72;                                 // xs2 row pad (bf16)

// ---------------------------------------------------------------------------
// Pack weights as Bp[kc][n][kk]  (kk contiguous, 32 per row) in bf16.
// k = kc*32+kk -> tap = kc>>1 (dy,dx), cin = (kc&1)*32 + kk; n = 64*g + cout.
// ---------------------------------------------------------------------------
__global__ void k_pack_w(const float* __restrict__ Ws,
                         const float* __restrict__ Wx,
                         const float* __restrict__ Wy,
                         __bf16* __restrict__ Bp) {
    int i = blockIdx.x * blockDim.x + threadIdx.x;
    const int total = NCH * NT * KC;
    if (i >= total) return;
    int kc = i / (NT * KC);
    int r  = i % (NT * KC);
    int n  = r / KC, kk = r % KC;
    int tap = kc >> 1;
    int dy = tap / 3, dx = tap % 3;
    int cin = (kc & 1) * 32 + kk;
    int g = n / 64, cout = n % 64;
    const float* Wg = (g == 0) ? Ws : ((g == 1) ? Wx : Wy);
    Bp[i] = (__bf16)Wg[cout * Ksum + cin * 9 + dy * 3 + dx];
}

// ---------------------------------------------------------------------------
// Fused 3-conv implicit GEMM, bf16 WMMA (f32 accumulate).
// Block: 384 threads = 12 waves; one 16-pixel strip (b,h,w0); wave w owns
// N columns [16w,16w+16). A fragments come straight from a cin-contiguous
// transposed strip (built once); B chunks are double-buffered in LDS with
// one barrier per chunk. All fragment traffic is ds_load_b128.
// ---------------------------------------------------------------------------
__global__ void __launch_bounds__(384)
k_conv(const float* __restrict__ x, const __bf16* __restrict__ Bp,
       float* __restrict__ outs,
       const float* __restrict__ bs, const float* __restrict__ bx,
       const float* __restrict__ by) {
    __shared__ __bf16 xs [Cc * 3 * 20];       // coalesced halo [cin][dy][wl]
    __shared__ __bf16 xs2[3 * 18 * XS2R];     // transposed     [dy][wl][cin]
    __shared__ __bf16 bsh[2][NT * BPAD];      // B chunk double buffer [n][kk]

    int bid = blockIdx.x;
    int b   = bid / (Hh * 8);
    int rem = bid % (Hh * 8);
    int h   = rem >> 3;
    int w0  = (rem & 7) << 4;
    int tid = threadIdx.x;

    // ---- stage x halo strip coalesced (zero padded), bf16 ----
    // 192 rows (cin,dy) x 18 cols; 2 threads per row, 9 consecutive cols each.
    {
        int row  = tid >> 1;                  // 0..191
        int half = tid & 1;
        int cin = row / 3, dy = row % 3;
        int y = h - 1 + dy;
        const float* xr = x + ((long long)b * Cc + cin) * PLANE + y * Ww;
#pragma unroll
        for (int i = 0; i < 9; ++i) {
            int wl = half * 9 + i;
            int w  = w0 - 1 + wl;
            float v = 0.0f;
            if ((unsigned)y < (unsigned)Hh && (unsigned)w < (unsigned)Ww)
                v = xr[w];
            xs[(cin * 3 + dy) * 20 + wl] = (__bf16)v;
        }
    }

    // B stage copy assignment (2 threads per n-row, 32B each)
    int brow = tid >> 1, bhalf = tid & 1;

    // ---- stage B chunk 0 into buffer 0 (independent of xs) ----
    {
        const uint4* g = (const uint4*)(Bp + (long long)brow * KC) + bhalf * 2;
        uint4* s = (uint4*)&bsh[0][brow * BPAD + bhalf * 16];
        s[0] = g[0];
        s[1] = g[1];
    }
    __syncthreads();

    // ---- one-time transpose: xs[cin][dy][wl] -> xs2[dy][wl][cin] ----
    for (int e = tid; e < 3 * 18 * 64; e += 384) {
        int cin = e & 63;
        int rc  = e >> 6;                     // 0..53 = dy*18+wl
        int dy  = rc / 18, wl = rc % 18;
        xs2[rc * XS2R + cin] = xs[(cin * 3 + dy) * 20 + wl];
    }
    __syncthreads();

    int wave = tid >> 5;
    int lane = tid & 31;
    int m    = lane & 15;                     // A row / B col / D col
    int hi   = lane >> 4;                     // lane half
    int nb   = wave * 16;                     // N base

    union Frag { uint4 u[2]; v16bf v; };

    v8f acc = {};
#pragma unroll
    for (int kc = 0; kc < NCH; ++kc) {
        // ---- issue next B chunk's global loads early ----
        uint4 g0, g1;
        if (kc + 1 < NCH) {
            const uint4* g =
                (const uint4*)(Bp + ((long long)(kc + 1) * NT + brow) * KC) + bhalf * 2;
            g0 = g[0];
            g1 = g[1];
        }
        if (kc + 2 < NCH)
            __builtin_prefetch((const void*)(Bp + (long long)(kc + 2) * NT * KC), 0, 0);

        // ---- fragments: 4x ds_load_b128 per lane ----
        int tap  = kc >> 1;
        int dy   = tap / 3, dx = tap % 3;
        int cinb = (kc & 1) * 32;
        int arow = (dy * 18 + m + dx) * XS2R;

        Frag af, bfr;
        af.u[0]  = *(const uint4*)&xs2[arow + cinb + hi * 8];        // K hi*8+0..7
        af.u[1]  = *(const uint4*)&xs2[arow + cinb + 16 + hi * 8];   // K 16+hi*8..
        bfr.u[0] = *(const uint4*)&bsh[kc & 1][(nb + m) * BPAD + hi * 8];
        bfr.u[1] = *(const uint4*)&bsh[kc & 1][(nb + m) * BPAD + 16 + hi * 8];

        acc = __builtin_amdgcn_wmma_f32_16x16x32_bf16(
            /*neg_a=*/false, af.v, /*neg_b=*/false, bfr.v,
            /*c_mod=*/(short)0, acc, /*reuse_a=*/false, /*reuse_b=*/false);

        // ---- store next chunk into the other buffer, single barrier ----
        if (kc + 1 < NCH) {
            uint4* s = (uint4*)&bsh[(kc + 1) & 1][brow * BPAD + bhalf * 16];
            s[0] = g0;
            s[1] = g1;
            __syncthreads();
        }
    }

    // ---- epilogue: D has N=m (channel), M=r+8*hi (pixel) -> 8 consecutive
    // floats per lane -> 2x global_store_b128 ----
    int g     = nb / 64;
    int coutb = nb % 64;
    const float* bias = (g == 0) ? bs : ((g == 1) ? bx : by);
    float bv = bias[coutb + m];
    float* og = outs + (long long)g * SZ;
    long long base = ((((long long)b * 64 + coutb + m) * Hh) + h) * Ww + w0 + hi * 8;
    float4 v0 = make_float4(acc[0] + bv, acc[1] + bv, acc[2] + bv, acc[3] + bv);
    float4 v1 = make_float4(acc[4] + bv, acc[5] + bv, acc[6] + bv, acc[7] + bv);
    *(float4*)(og + base)     = v0;
    *(float4*)(og + base + 4) = v1;
}

// ---------------------------------------------------------------------------
// Bilinear grid-sample (torch bilinear/zeros/align_corners=False), fp32.
// ---------------------------------------------------------------------------
__global__ void k_sample(const float* __restrict__ wsf, float* __restrict__ out) {
    const float* score = wsf;
    const float* gx    = wsf + SZ;
    const float* gy    = wsf + 2 * SZ;
    long long idx = (long long)blockIdx.x * blockDim.x + threadIdx.x;
    if (idx >= SZ) return;
    long long plane = idx / PLANE;
    const float* img = score + plane * PLANE;
    float fx = gx[idx], fy = gy[idx];
    float ix = ((fx + 1.0f) * (float)Ww - 1.0f) * 0.5f;
    float iy = ((fy + 1.0f) * (float)Hh - 1.0f) * 0.5f;
    float x0f = floorf(ix), y0f = floorf(iy);
    int x0 = (int)x0f, y0 = (int)y0f;
    int x1 = x0 + 1, y1 = y0 + 1;
    float wx1 = ix - x0f, wy1 = iy - y0f;
    float wx0 = 1.0f - wx1, wy0 = 1.0f - wy1;

    auto gather = [&](int yi, int xi) -> float {
        if ((unsigned)yi < (unsigned)Hh && (unsigned)xi < (unsigned)Ww)
            return img[yi * Ww + xi];
        return 0.0f;
    };
    out[idx] = gather(y0, x0) * (wy0 * wx0)
             + gather(y0, x1) * (wy0 * wx1)
             + gather(y1, x0) * (wy1 * wx0)
             + gather(y1, x1) * (wy1 * wx1);
}

// ---------------------------------------------------------------------------
extern "C" void kernel_launch(void* const* d_in, const int* in_sizes, int n_in,
                              void* d_out, int out_size, void* d_ws, size_t ws_size,
                              hipStream_t stream) {
    (void)in_sizes; (void)n_in; (void)out_size; (void)ws_size;
    const float* x  = (const float*)d_in[0];
    const float* Ws = (const float*)d_in[1];
    const float* bs = (const float*)d_in[2];
    const float* Wx = (const float*)d_in[3];
    const float* bx = (const float*)d_in[4];
    const float* Wy = (const float*)d_in[5];
    const float* by = (const float*)d_in[6];

    // Workspace: [score | gx | gy] f32 (3 * 64 MiB), then packed bf16 B.
    float*  wsf = (float*)d_ws;
    __bf16* Bp  = (__bf16*)((char*)d_ws + (size_t)(3 * SZ) * sizeof(float));

    {
        int total = NCH * NT * KC;
        k_pack_w<<<(total + 255) / 256, 256, 0, stream>>>(Ws, Wx, Wy, Bp);
    }
    {
        int grid = Bn * Hh * (Ww / 16);   // 16384 blocks, 384 threads (12 waves)
        k_conv<<<grid, 384, 0, stream>>>(x, Bp, wsf, bs, bx, by);
    }
    {
        long long total = SZ;
        k_sample<<<(int)((total + 255) / 256), 256, 0, stream>>>(wsf, (float*)d_out);
    }
}
